// VGG16RoIHead_9663676416509
// MI455X (gfx1250) — matrix operand order repaired and checked by reading, not compiled, not run on hardware
//
#include <hip/hip_runtime.h>

// ---------------------------------------------------------------------------
// VGG16 RoI head on MI455X (gfx1250, wave32):
//   pool  = RoIAdaptiveMaxPool(x, rois)          [300, 25088]
//   fc6   = relu(pool @ W6 + b6)                 [300, 4096]
//   fc7   = relu(fc6  @ W7 + b7)                 [300, 4096]
//   locs  = fc7 @ Wl + bl                        [300, 84]  -> d_out[0:25200]
//   score = fc7 @ Ws + bs                        [300, 21]  -> d_out[25200:]
//
// GEMMs use V_WMMA_F32_16X16X32_BF16 (f32 accumulate). BOTH operands are kept
// in pre-packed WMMA fragment layout in global memory:
//   A-frags: Ap[((mt*KT + kt)*32 + lane)*16 + j]  (16x32 tile per (mt,kt))
//   B-frags: Bp[((nt*KT + kt)*32 + lane)*16 + j]  (32x16 tile per (nt,kt))
// The GEMM inner loop is pure coalesced global b128 loads + WMMA: no LDS, no
// barriers, explicit two-set register double buffering (KT must be even; all
// packed buffers carry one extra tile so the pipeline's kt+2 load needs no
// guard).
// ---------------------------------------------------------------------------

typedef __attribute__((ext_vector_type(16))) __bf16 v16bf;
typedef __attribute__((ext_vector_type(8)))  __bf16 v8bf;
typedef __attribute__((ext_vector_type(8)))  float  v8f;

#define C_CH   512
#define H_F    50
#define W_F    50
#define R_ROI  300
#define POOLED (C_CH * 49)      // 25088
#define NHID   4096
#define NLOC   84
#define NSCORE 21

// round-to-nearest-even f32 -> bf16 (stored as ushort)
__device__ __forceinline__ unsigned short f2bf(float f) {
  union { float f; unsigned u; } v; v.f = f;
  unsigned u = v.u;
  unsigned r = u + 0x7FFFu + ((u >> 16) & 1u);
  return (unsigned short)(r >> 16);
}

// Scatter one bf16 value into packed A-fragment layout.
// A 16x32 tile layout (contiguous, matching packed B): lanes 0-15: M=lane,
// j=0..15 -> K 0..15; lanes 16-31: M=lane-16, j=0..15 -> K 16..31.
// Inverse: koff=k&31 -> lane_hi=(koff>>4)&1,
// j=koff&15.
__device__ __forceinline__ void store_packedA(unsigned short* __restrict__ P,
                                              int KTp, int m, int k,
                                              unsigned short v) {
  int mt = m >> 4, mlane = m & 15;
  int kt = k >> 5, koff = k & 31;
  int lane_hi = (koff >> 4) & 1;
  int j = koff & 15;
  P[((((size_t)mt * KTp + kt) * 32) + lane_hi * 16 + mlane) * 16 + j] = v;
}

// ---------------------------------------------------------------------------
// Pack W[K][N] (f32 row-major) into WMMA B-fragment layout (bf16):
//   Bp[((nt*KT + kt)*32 + lane)*16 + j] = bf16(W[kt*32 + kb(lane) + j][nt*16 + (lane&15)])
// kb(lane) = (lane>=16) ? 16 : 0.  One thread per (nt,kt,lane).
// ---------------------------------------------------------------------------
__global__ __launch_bounds__(256)
void pack_b_bf16(const float* __restrict__ W, unsigned short* __restrict__ Bp,
                 int N, int KT, int NT) {
  size_t t = (size_t)blockIdx.x * blockDim.x + threadIdx.x;
  size_t total = (size_t)NT * KT * 32;
  if (t >= total) return;
  int lane = (int)(t & 31);
  size_t tt = t >> 5;
  int kt = (int)(tt % KT);
  int nt = (int)(tt / KT);
  int n = min(nt * 16 + (lane & 15), N - 1);          // clamp padded columns
  int kbase = kt * 32 + ((lane >> 4) ? 16 : 0);
  unsigned short tmp[16];
#pragma unroll
  for (int j = 0; j < 16; ++j)
    tmp[j] = f2bf(W[(size_t)(kbase + j) * N + n]);
  unsigned w[8];
#pragma unroll
  for (int j = 0; j < 8; ++j)
    w[j] = (unsigned)tmp[2 * j] | ((unsigned)tmp[2 * j + 1] << 16);
  uint4* dst = (uint4*)(Bp + t * 16);                 // 32B-aligned
  dst[0] = make_uint4(w[0], w[1], w[2], w[3]);
  dst[1] = make_uint4(w[4], w[5], w[6], w[7]);
}

// ---------------------------------------------------------------------------
// RoI adaptive max-pool, writing bf16 directly in packed A-fragment layout.
// One block per ROI.  Matches reference bin math exactly.
// ---------------------------------------------------------------------------
__global__ __launch_bounds__(256)
void roipool_packA_bf16(const float* __restrict__ x,     // [512,50,50]
                        const float* __restrict__ rois,  // [300,4] (y1,x1,y2,x2)
                        unsigned short* __restrict__ Ap) {
  int r = blockIdx.x;
  const float* roi = rois + r * 4;
  int y1 = (int)(roi[0] * 0.0625f);
  int x1 = (int)(roi[1] * 0.0625f);
  int y2 = (int)(roi[2] * 0.0625f);
  int x2 = (int)(roi[3] * 0.0625f);
  int h = y2 - y1 + 1;
  int w = x2 - x1 + 1;
  const int KT6 = POOLED / 32;
  for (int e = threadIdx.x; e < POOLED; e += blockDim.x) {
    int c = e / 49, rem = e % 49, i = rem / 7, j = rem % 7;
    int rs = y1 + (i * h) / 7;
    int re = y1 + ((i + 1) * h + 6) / 7;   // ceil
    int cs = x1 + (j * w) / 7;
    int ce = x1 + ((j + 1) * w + 6) / 7;   // ceil
    const float* fc = x + (size_t)c * (H_F * W_F);
    float m = -3.402823466e+38f;
    for (int yy = rs; yy < re; ++yy)
      for (int xx = cs; xx < ce; ++xx)
        m = fmaxf(m, fc[yy * W_F + xx]);
    store_packedA(Ap, KT6, r, e, f2bf(m));
  }
}

// ---------------------------------------------------------------------------
// bf16 WMMA GEMM with both operands pre-packed in fragment layout.
// Block: 256 threads = 8 waves; block tile M=16, N=256.
// Each wave: two 16x16 subtiles sharing one A fragment.  No LDS, no barriers.
// Explicit 2-deep register double buffering; KT must be even and >= 2, and
// packed buffers must have one spare tile at the end (for the kt+2 load).
// Epilogue: bias (+ReLU); writes f32 [M,N] and/or bf16 in packed-A layout
// (KTout = N/32) for chaining into the next GEMM.
// ---------------------------------------------------------------------------
__global__ __launch_bounds__(256)
void gemm_wmma_bf16(const unsigned short* __restrict__ Ap,  // packed [MT][KT][32][16]
                    const unsigned short* __restrict__ Bp,  // packed [NT][KT][32][16]
                    const float* __restrict__ bias,         // [N]
                    unsigned short* __restrict__ outp,      // packed-A bf16 out or null
                    float* __restrict__ outf,               // [M,N] f32 out or null
                    int M, int N, int K, int NT, int relu) {
  const int KT = K >> 5;                 // even, >= 2
  const int mt   = blockIdx.y;
  const int tid  = threadIdx.x;
  const int lane = tid & 31;
  const int wave = tid >> 5;
  const int nt0  = blockIdx.x * 16 + wave * 2;
  const int nt1  = nt0 + 1;
  const int nt0c = min(nt0, NT - 1);     // clamp for padded N tiles
  const int nt1c = min(nt1, NT - 1);

  const unsigned short* pA  = Ap + ((size_t)mt   * KT * 32 + lane) * 16;
  const unsigned short* pB0 = Bp + ((size_t)nt0c * KT * 32 + lane) * 16;
  const unsigned short* pB1 = Bp + ((size_t)nt1c * KT * 32 + lane) * 16;

  v8f acc0 = {};
  v8f acc1 = {};

#define LDFRAG(base, kt_, lo, hi)                                   \
  {                                                                 \
    const v8bf* _p = (const v8bf*)((base) + (size_t)(kt_) * 512);   \
    lo = _p[0];                                                     \
    hi = _p[1];                                                     \
  }
#define MKV16(lo, hi)                                               \
  __builtin_shufflevector(lo, hi, 0, 1, 2, 3, 4, 5, 6, 7, 8, 9,     \
                          10, 11, 12, 13, 14, 15)
#define WMMA(a, b, c)                                               \
  __builtin_amdgcn_wmma_f32_16x16x32_bf16(false, a, false, b,       \
                                          (short)0, c, false, false)

  // two explicit register sets: a (A frag), b/c (two B subtile frags)
  v8bf a0l, a0h, b0l, b0h, c0l, c0h;     // set 0
  v8bf a1l, a1h, b1l, b1h, c1l, c1h;     // set 1

  LDFRAG(pA,  0, a0l, a0h);
  LDFRAG(pB0, 0, b0l, b0h);
  LDFRAG(pB1, 0, c0l, c0h);

  for (int kt = 0; kt < KT; kt += 2) {
    // stage kt+1 into set 1 (always in-bounds: KT even)
    LDFRAG(pA,  kt + 1, a1l, a1h);
    LDFRAG(pB0, kt + 1, b1l, b1h);
    LDFRAG(pB1, kt + 1, c1l, c1h);
    __builtin_prefetch(pB0 + (size_t)(kt + 8) * 512, 0, 1);  // speculative
    __builtin_prefetch(pB1 + (size_t)(kt + 8) * 512, 0, 1);

    acc0 = WMMA(MKV16(a0l, a0h), MKV16(b0l, b0h), acc0);
    acc1 = WMMA(MKV16(a0l, a0h), MKV16(c0l, c0h), acc1);

    // stage kt+2 into set 0 (last iteration reads the spare tile)
    LDFRAG(pA,  kt + 2, a0l, a0h);
    LDFRAG(pB0, kt + 2, b0l, b0h);
    LDFRAG(pB1, kt + 2, c0l, c0h);

    acc0 = WMMA(MKV16(a1l, a1h), MKV16(b1l, b1h), acc0);
    acc1 = WMMA(MKV16(a1l, a1h), MKV16(c1l, c1h), acc1);
  }
#undef LDFRAG
#undef MKV16
#undef WMMA

  // ---- epilogue: bias + relu; store f32 row-major and/or packed-A bf16
  // C/D layout: VGPR r -> row = mt*16 + (lane>=16 ? 8 : 0) + r, col = lane&15
  const int rbase = mt * 16 + (lane >> 4) * 8;
  const int col0  = nt0 * 16 + (lane & 15);
  const int col1  = nt1 * 16 + (lane & 15);
  const float bia0 = (col0 < N) ? bias[col0] : 0.0f;
  const float bia1 = (col1 < N) ? bias[col1] : 0.0f;
  const int KTout = N >> 5;   // packed-A KT of the chained output
#pragma unroll
  for (int r = 0; r < 8; ++r) {
    int rr = rbase + r;
    if (rr >= M) continue;
    float v0 = acc0[r] + bia0;
    float v1 = acc1[r] + bia1;
    if (relu) { v0 = fmaxf(v0, 0.0f); v1 = fmaxf(v1, 0.0f); }
    if (col0 < N) {
      if (outf) outf[(size_t)rr * N + col0] = v0;
      if (outp) store_packedA(outp, KTout, rr, col0, f2bf(v0));
    }
    if (col1 < N) {
      if (outf) outf[(size_t)rr * N + col1] = v1;
      if (outp) store_packedA(outp, KTout, rr, col1, f2bf(v1));
    }
  }
}

// ---------------------------------------------------------------------------
// launch
// ---------------------------------------------------------------------------
extern "C" void kernel_launch(void* const* d_in, const int* in_sizes, int n_in,
                              void* d_out, int out_size, void* d_ws, size_t ws_size,
                              hipStream_t stream) {
  const float* x       = (const float*)d_in[0];
  const float* rois    = (const float*)d_in[1];
  const float* W_fc6   = (const float*)d_in[2];
  const float* b_fc6   = (const float*)d_in[3];
  const float* W_fc7   = (const float*)d_in[4];
  const float* b_fc7   = (const float*)d_in[5];
  const float* W_loc   = (const float*)d_in[6];
  const float* b_loc   = (const float*)d_in[7];
  const float* W_score = (const float*)d_in[8];
  const float* b_score = (const float*)d_in[9];
  float* out = (float*)d_out;

  // tile counts
  const int KT6 = POOLED / 32;              // 784 (even)
  const int KT7 = NHID / 32;                // 128 (even)
  const int NTH = NHID / 16;                // 256
  const int NTL = (NLOC + 15) / 16;         // 6
  const int NTS = (NSCORE + 15) / 16;       // 2
  const int MT  = (R_ROI + 15) / 16;        // 19

  // carve workspace (256B-aligned); packed tiles = 512 halves (1KB) each.
  // Every packed operand buffer gets ONE SPARE TILE for the pipeline's
  // unguarded kt+2 load.
  size_t off = 0;
  auto carve = [&](size_t bytes) -> unsigned short* {
    unsigned short* p = (unsigned short*)((char*)d_ws + off);
    off += (bytes + 255) & ~(size_t)255;
    return p;
  };
  const size_t SPARE = 512 * 2;                                       // 1KB
  unsigned short* A6p = carve((size_t)MT  * KT6 * 512 * 2 + SPARE);   // pool, packed A
  unsigned short* W6p = carve((size_t)NTH * KT6 * 512 * 2 + SPARE);   // 205 MB
  unsigned short* W7p = carve((size_t)NTH * KT7 * 512 * 2 + SPARE);   // 33.5 MB
  unsigned short* WLp = carve((size_t)NTL * KT7 * 512 * 2 + SPARE);
  unsigned short* WSp = carve((size_t)NTS * KT7 * 512 * 2 + SPARE);
  unsigned short* H6p = carve((size_t)MT  * KT7 * 512 * 2 + SPARE);   // fc6 out, packed A
  unsigned short* H7p = carve((size_t)MT  * KT7 * 512 * 2 + SPARE);   // fc7 out, packed A

  // 1) convert + pack weights into WMMA B-fragment layout
  {
    size_t t6 = (size_t)NTH * KT6 * 32;
    pack_b_bf16<<<(unsigned)((t6 + 255) / 256), 256, 0, stream>>>(W_fc6, W6p, NHID, KT6, NTH);
    size_t t7 = (size_t)NTH * KT7 * 32;
    pack_b_bf16<<<(unsigned)((t7 + 255) / 256), 256, 0, stream>>>(W_fc7, W7p, NHID, KT7, NTH);
    size_t tl = (size_t)NTL * KT7 * 32;
    pack_b_bf16<<<(unsigned)((tl + 255) / 256), 256, 0, stream>>>(W_loc, WLp, NLOC, KT7, NTL);
    size_t ts = (size_t)NTS * KT7 * 32;
    pack_b_bf16<<<(unsigned)((ts + 255) / 256), 256, 0, stream>>>(W_score, WSp, NSCORE, KT7, NTS);
  }

  // 2) RoI adaptive max pool -> packed-A bf16
  roipool_packA_bf16<<<R_ROI, 256, 0, stream>>>(x, rois, A6p);

  // 3) fc6 = relu(pool @ W6 + b6) -> H6 (packed A)
  gemm_wmma_bf16<<<dim3(NTH / 16, MT), 256, 0, stream>>>(
      A6p, W6p, b_fc6, H6p, nullptr, R_ROI, NHID, POOLED, NTH, 1);

  // 4) fc7 = relu(H6 @ W7 + b7) -> H7 (packed A)
  gemm_wmma_bf16<<<dim3(NTH / 16, MT), 256, 0, stream>>>(
      H6p, W7p, b_fc7, H7p, nullptr, R_ROI, NHID, NHID, NTH, 1);

  // 5) roi_cls_locs = H7 @ WL + bl -> d_out[0 : 300*84] (f32)
  gemm_wmma_bf16<<<dim3((NTL + 15) / 16, MT), 256, 0, stream>>>(
      H7p, WLp, b_loc, nullptr, out, R_ROI, NLOC, NHID, NTL, 0);

  // 6) roi_scores = H7 @ WS + bs -> d_out[300*84 : ] (f32)
  gemm_wmma_bf16<<<dim3((NTS + 15) / 16, MT), 256, 0, stream>>>(
      H7p, WSp, b_score, nullptr, out + (size_t)R_ROI * NLOC, R_ROI, NSCORE, NHID, NTS, 0);
}
